// MultiHeadAttnCoupling_55018531062690
// MI455X (gfx1250) — matrix operand order repaired and checked by reading, not compiled, not run on hardware
//
#include <hip/hip_runtime.h>
#include <hip/hip_bf16.h>

typedef __attribute__((ext_vector_type(16))) __bf16 bf16x16;
typedef __attribute__((ext_vector_type(8)))  float  f32x8;

#define DD 32768   // D = heads(8) * seq(64) * d_tensor(64)
#define TOK 512    // B*N tokens

// Pack two f32 -> two bf16 in one 32-bit word (low = a, high = b).
// Single V_CVT_PK_BF16_F32 (RNE) — the clang builtin is absent on this
// toolchain, so fall back to inline asm for the same instruction.
static __device__ __forceinline__ unsigned int cvt2_bf16(float a, float b) {
#if __has_builtin(__builtin_amdgcn_cvt_pk_bf16_f32)
  auto p = __builtin_amdgcn_cvt_pk_bf16_f32(a, b);
  union { decltype(p) v; unsigned int u; } cv; cv.v = p;
  return cv.u;
#else
  unsigned int r;
  asm("v_cvt_pk_bf16_f32 %0, %1, %2" : "=v"(r) : "v"(a), "v"(b));
  return r;
#endif
}
static __device__ __forceinline__ unsigned short f32_to_bf16(float f) {
  return (unsigned short)cvt2_bf16(f, 0.0f);
}

struct FragBF {
  union { bf16x16 v; uint4 q[2]; };
};

// Async copy 16B global -> LDS (ASYNCcnt-tracked, no VGPR data round-trip).
// VDST holds the LDS byte offset (low 32 bits of a generic LDS pointer).
static __device__ __forceinline__ void async_copy_b128(void* lds_ptr, const void* gptr) {
  unsigned lds = (unsigned)(unsigned long long)lds_ptr;
  asm volatile("global_load_async_to_lds_b128 %0, %1, off"
               :: "v"(lds), "v"(gptr)
               : "memory");
}
static __device__ __forceinline__ void wait_async0() {
  asm volatile("s_wait_asynccnt 0" ::: "memory");
}

// A-matrix fragment (16x32 bf16): lane m = lane&15, half = lane>>4.
// VGPR e: k = 16*(e>=8) + 8*half + (e&7)  ->  two contiguous 16B chunks.
static __device__ __forceinline__ void frag_a(const unsigned short* lds, int row0,
                                              int pitch, int k0, FragBF& f) {
  const int lane = threadIdx.x & 31;
  const int half = lane >> 4;
  const int r    = lane & 15;
  const unsigned short* p = lds + (row0 + r) * pitch + k0 + half * 8;
  f.q[0] = *(const uint4*)(p);
  f.q[1] = *(const uint4*)(p + 16);
}

// B-matrix fragment (32x16 bf16): lane n = lane&15, k = 16*(e>=8) + 8*half + (e&7) -> two 16B chunks.
static __device__ __forceinline__ void frag_b(const unsigned short* lds, int n0,
                                              int pitch, int k0, FragBF& f) {
  const int lane = threadIdx.x & 31;
  const int half = lane >> 4;
  const int n    = lane & 15;
  const unsigned short* p = lds + (n0 + n) * pitch + k0 + half * 8;
  f.q[0] = *(const uint4*)(p);
  f.q[1] = *(const uint4*)(p + 16);
}

static __device__ __forceinline__ f32x8 wmma_bf16(const FragBF& a, const FragBF& b, f32x8 c) {
  return __builtin_amdgcn_wmma_f32_16x16x32_bf16(false, a.v, false, b.v, (short)0, c, false, false);
}

// Transposed B staging: one float4 (row k, cols 4c..4c+3) -> 4 LDS rows, 1 column.
// Packed cvt + lo/hi b16 stores (compiler emits ds_store_b16[_d16_hi]).
static __device__ __forceinline__ void stage_b4(unsigned short* sB, int r, int c,
                                                const float4& w) {
  unsigned int p0 = cvt2_bf16(w.x, w.y);
  unsigned int p1 = cvt2_bf16(w.z, w.w);
  unsigned short* bp = sB + (c * 4) * 72 + r;
  bp[0]       = (unsigned short)p0;
  bp[72]      = (unsigned short)(p0 >> 16);
  bp[144]     = (unsigned short)p1;
  bp[216]     = (unsigned short)(p1 >> 16);
}

// ---------------------------------------------------------------------------
// Kernel 1: QKV projections.  [512 x Kd] @ [Kd x 32768] + bias -> bf16.
// grid = (256 n-blocks of 128, 8 m-blocks of 64, 3 = {Q,K,V}), block = 256 (8 waves).
// Register double-buffered: next K-tile global loads overlap WMMA compute.
// V is stored transposed per head tile: [token][h][d][seq].
// ---------------------------------------------------------------------------
__global__ __launch_bounds__(256) void qkv_gemm_kernel(
    const float* __restrict__ x, const float* __restrict__ z,
    const float* __restrict__ Wq, const float* __restrict__ bq,
    const float* __restrict__ Wk, const float* __restrict__ bk,
    const float* __restrict__ Wv, const float* __restrict__ bv,
    unsigned short* __restrict__ qbuf, unsigned short* __restrict__ kbuf,
    unsigned short* __restrict__ vtbuf)
{
  const int which = blockIdx.z;
  const float* A; const float* W; const float* bias; unsigned short* dst; int Kd;
  if (which == 0)      { A = z; W = Wq; bias = bq; dst = qbuf;  Kd = 256; }
  else if (which == 1) { A = x; W = Wk; bias = bk; dst = kbuf;  Kd = 512; }
  else                 { A = x; W = Wv; bias = bv; dst = vtbuf; Kd = 512; }

  const int n0g = blockIdx.x * 128;
  const int m0g = blockIdx.y * 64;

  __shared__ unsigned short sA[64 * 72];    // [m][k], pitch 72 (16B aligned, conflict-free)
  __shared__ unsigned short sB[128 * 72];   // [n][k] (transposed during staging)

  const int tid  = threadIdx.x;
  const int wave = tid >> 5, lane = tid & 31, half = lane >> 4, lc = lane & 15;
  const int wm = wave & 1, wn = wave >> 1;  // 2 x 4 wave grid -> each wave 32x32 output

  // Staging index precompute.
  const int rA0 = tid >> 3,          cA0 = tid & 7;          // A: 2 float4/thread
  const int rA1 = (tid + 256) >> 3,  cA1 = (tid + 256) & 7;
  int rB[4], cB[4];
  for (int i = 0; i < 4; ++i) { int e4 = tid + i * 256; rB[i] = e4 >> 5; cB[i] = e4 & 31; }

  const f32x8 zero = {0.f,0.f,0.f,0.f,0.f,0.f,0.f,0.f};
  f32x8 acc[2][2] = {{zero, zero}, {zero, zero}};

  float4 ra0, ra1, rb[4];
  auto g_load = [&](int k0) {
    ra0 = *(const float4*)(A + (size_t)(m0g + rA0) * Kd + k0 + cA0 * 4);
    ra1 = *(const float4*)(A + (size_t)(m0g + rA1) * Kd + k0 + cA1 * 4);
    for (int i = 0; i < 4; ++i)
      rb[i] = *(const float4*)(W + (size_t)(k0 + rB[i]) * DD + n0g + cB[i] * 4);
  };

  g_load(0);
  for (int k0 = 0; k0 < Kd; k0 += 32) {
    // Stage current tile (regs -> LDS, f32 -> bf16 via packed cvt).
    *(uint2*)(sA + rA0 * 72 + cA0 * 4) = make_uint2(cvt2_bf16(ra0.x, ra0.y), cvt2_bf16(ra0.z, ra0.w));
    *(uint2*)(sA + rA1 * 72 + cA1 * 4) = make_uint2(cvt2_bf16(ra1.x, ra1.y), cvt2_bf16(ra1.z, ra1.w));
    for (int i = 0; i < 4; ++i) stage_b4(sB, rB[i], cB[i], rb[i]);
    __syncthreads();
    if (k0 + 32 < Kd) g_load(k0 + 32);   // in-flight during WMMA below

    FragBF a0, a1, b0, b1;
    frag_a(sA, wm * 32 +  0, 72, 0, a0);
    frag_a(sA, wm * 32 + 16, 72, 0, a1);
    frag_b(sB, wn * 32 +  0, 72, 0, b0);
    frag_b(sB, wn * 32 + 16, 72, 0, b1);
    acc[0][0] = wmma_bf16(a0, b0, acc[0][0]);
    acc[0][1] = wmma_bf16(a0, b1, acc[0][1]);
    acc[1][0] = wmma_bf16(a1, b0, acc[1][0]);
    acc[1][1] = wmma_bf16(a1, b1, acc[1][1]);
    __syncthreads();
  }

  // Epilogue: +bias, packed f32 -> bf16, store (V transposed per head).
  for (int mi = 0; mi < 2; ++mi)
    for (int ni = 0; ni < 2; ++ni) {
      int ng = n0g + wn * 32 + ni * 16 + lc;
      float bb = bias[ng];
      for (int i = 0; i < 8; i += 2) {
        int mg = m0g + wm * 32 + mi * 16 + half * 8 + i;
        unsigned int p = cvt2_bf16(acc[mi][ni][i] + bb, acc[mi][ni][i + 1] + bb);
        size_t idx0, idx1;
        if (which == 2) {
          int hh = ng >> 12, ss = (ng >> 6) & 63, dv = ng & 63;
          size_t col = ((size_t)hh << 12) + dv * 64 + ss;
          idx0 = (size_t)mg * DD + col;
          idx1 = idx0 + DD;                 // mg+1, same transposed column
        } else {
          idx0 = (size_t)mg * DD + ng;
          idx1 = idx0 + DD;                 // mg+1, same column
        }
        dst[idx0] = (unsigned short)p;
        dst[idx1] = (unsigned short)(p >> 16);
      }
    }
}

// ---------------------------------------------------------------------------
// Kernel 2: per-(b,n,head) attention.  4096 blocks, 256 threads (8 waves).
// Tiles staged with ASYNC global->LDS copies.  Output aliases qbuf slice.
// ---------------------------------------------------------------------------
__global__ __launch_bounds__(256) void attn_kernel(
    const unsigned short* __restrict__ qbuf, const unsigned short* __restrict__ kbuf,
    const unsigned short* __restrict__ vtbuf, unsigned short* __restrict__ obuf)
{
  const size_t base = (size_t)blockIdx.x * 4096;   // 64x64 bf16 tile per head
  __shared__ unsigned short sQ[64 * 72];   // [q][d]
  __shared__ unsigned short sK[64 * 72];   // [kpos][d]
  __shared__ unsigned short sV[64 * 72];   // [d][kpos]  (V already transposed)
  __shared__ unsigned short sP[64 * 72];   // softmaxed probs [q][kpos], bf16
  __shared__ float          sS[64 * 65];   // raw scores f32

  const int tid  = threadIdx.x;
  const int wave = tid >> 5, lane = tid & 31, half = lane >> 4, lc = lane & 15;
  const f32x8 zero = {0.f,0.f,0.f,0.f,0.f,0.f,0.f,0.f};

  // Stage Q, K, Vt tiles via async LDS copies: 512 uint4 each, 2 per thread per tile.
  for (int i = 0; i < 2; ++i) {
    int e4 = tid + i * 256;
    int r = e4 >> 3, c8 = e4 & 7;
    async_copy_b128(sQ + r * 72 + c8 * 8, qbuf  + base + r * 64 + c8 * 8);
    async_copy_b128(sK + r * 72 + c8 * 8, kbuf  + base + r * 64 + c8 * 8);
    async_copy_b128(sV + r * 72 + c8 * 8, vtbuf + base + r * 64 + c8 * 8);
  }
  wait_async0();
  __syncthreads();

  // scores: 4x4 tile grid, 2 tiles per wave.
  const int q0 = (wave & 3) * 16;
  const int c0 = (wave >> 2) * 32;
  {
    f32x8 acc[2] = {zero, zero};
    for (int k0 = 0; k0 < 64; k0 += 32) {
      FragBF a; frag_a(sQ, q0, 72, k0, a);
      for (int ni = 0; ni < 2; ++ni) {
        FragBF b; frag_b(sK, c0 + ni * 16, 72, k0, b);
        acc[ni] = wmma_bf16(a, b, acc[ni]);
      }
    }
    for (int ni = 0; ni < 2; ++ni)
      for (int i = 0; i < 8; ++i)
        sS[(q0 + half * 8 + i) * 65 + (c0 + ni * 16 + lc)] = acc[ni][i] * 0.125f;
  }
  __syncthreads();

  // Softmax: one thread per row; packed bf16 stores of the probabilities.
  if (tid < 64) {
    float* row = sS + tid * 65;
    float m = -3.4e38f;
    for (int c = 0; c < 64; ++c) m = fmaxf(m, row[c]);
    float s = 0.f;
    for (int c = 0; c < 64; ++c) { float e = __expf(row[c] - m); row[c] = e; s += e; }
    float inv = 1.f / s;
    for (int c = 0; c < 64; c += 2)
      *(unsigned int*)(sP + tid * 72 + c) = cvt2_bf16(row[c] * inv, row[c + 1] * inv);
  }
  __syncthreads();

  // out = P @ Vt : same 4x4 tiling; packed epilogue (rows q, q+1).
  {
    const int d0 = (wave >> 2) * 32;
    f32x8 acc[2] = {zero, zero};
    for (int k0 = 0; k0 < 64; k0 += 32) {
      FragBF a; frag_a(sP, q0, 72, k0, a);
      for (int ni = 0; ni < 2; ++ni) {
        FragBF b; frag_b(sV, d0 + ni * 16, 72, k0, b);
        acc[ni] = wmma_bf16(a, b, acc[ni]);
      }
    }
    for (int ni = 0; ni < 2; ++ni)
      for (int i = 0; i < 8; i += 2) {
        int q = q0 + half * 8 + i;
        int d = d0 + ni * 16 + lc;
        unsigned int p = cvt2_bf16(acc[ni][i], acc[ni][i + 1]);
        obuf[base + q * 64 + d]       = (unsigned short)p;
        obuf[base + (q + 1) * 64 + d] = (unsigned short)(p >> 16);
      }
  }
}

// ---------------------------------------------------------------------------
// Kernel 3: output projection, split-K.  [512 x 32768] @ [32768 x 512].
// grid = (8 n, 8 m, 8 k-chunks of 4096), block = 256.  f32 partials to ws.
// A tile: async global->LDS (already bf16).  B tile: register double-buffered
// f32->bf16 transpose.
// ---------------------------------------------------------------------------
__global__ __launch_bounds__(256) void out_proj_kernel(
    const unsigned short* __restrict__ abuf, const float* __restrict__ Wo,
    float* __restrict__ partial)
{
  const int n0g   = blockIdx.x * 64;
  const int m0g   = blockIdx.y * 64;
  const int kbase = blockIdx.z * 4096;

  __shared__ unsigned short sA[64 * 72];   // [m][k] bf16 (async direct copy)
  __shared__ unsigned short sB[64 * 72];   // [n][k] bf16 (transposed f32->bf16)

  const int tid  = threadIdx.x;
  const int wave = tid >> 5, lane = tid & 31, half = lane >> 4, lc = lane & 15;
  const int mi  = wave >> 1;
  const int ni0 = (wave & 1) * 2;

  const int rAT = tid >> 2, cAT = tid & 3;                    // A: 1 uint4/thread
  int rBT[2], cBT[2];
  for (int i = 0; i < 2; ++i) { int e4 = tid + i * 256; rBT[i] = e4 >> 4; cBT[i] = e4 & 15; }

  const f32x8 zero = {0.f,0.f,0.f,0.f,0.f,0.f,0.f,0.f};
  f32x8 acc[2] = {zero, zero};

  float4 rb[2];
  auto g_loadB = [&](int k0) {
    for (int i = 0; i < 2; ++i)
      rb[i] = *(const float4*)(Wo + (size_t)(kbase + k0 + rBT[i]) * 512 + n0g + cBT[i] * 4);
  };

  g_loadB(0);
  for (int k0 = 0; k0 < 4096; k0 += 32) {
    // A tile: async copy 64x32 bf16 (256 uint4, 1 per thread).
    async_copy_b128(sA + rAT * 72 + cAT * 8,
                    abuf + (size_t)(m0g + rAT) * DD + kbase + k0 + cAT * 8);
    // B tile: staged regs -> LDS transposed (packed cvt).
    for (int i = 0; i < 2; ++i) stage_b4(sB, rBT[i], cBT[i], rb[i]);
    wait_async0();
    __syncthreads();
    if (k0 + 32 < 4096) g_loadB(k0 + 32);   // overlap with WMMA below

    FragBF a; frag_a(sA, mi * 16, 72, 0, a);
    for (int j = 0; j < 2; ++j) {
      FragBF b; frag_b(sB, (ni0 + j) * 16, 72, 0, b);
      acc[j] = wmma_bf16(a, b, acc[j]);
    }
    __syncthreads();
  }

  float* p = partial + (size_t)blockIdx.z * (512 * 512);
  for (int j = 0; j < 2; ++j)
    for (int i = 0; i < 8; ++i) {
      int mg = m0g + mi * 16 + half * 8 + i;
      int ng = n0g + (ni0 + j) * 16 + lc;
      p[(size_t)mg * 512 + ng] = acc[j][i];
    }
}

// ---------------------------------------------------------------------------
// Kernel 4: reduce split-K partials + bias -> f32 output.
// ---------------------------------------------------------------------------
__global__ __launch_bounds__(256) void reduce_kernel(
    const float* __restrict__ partial, const float* __restrict__ bo,
    float* __restrict__ out)
{
  int idx = blockIdx.x * 256 + threadIdx.x;   // < 512*512
  float s = bo[idx & 511];
  for (int c = 0; c < 8; ++c) s += partial[(size_t)c * (512 * 512) + idx];
  out[idx] = s;
}

// ---------------------------------------------------------------------------
extern "C" void kernel_launch(void* const* d_in, const int* in_sizes, int n_in,
                              void* d_out, int out_size, void* d_ws, size_t ws_size,
                              hipStream_t stream) {
  (void)in_sizes; (void)n_in; (void)out_size; (void)ws_size;
  const float* x  = (const float*)d_in[0];
  const float* z  = (const float*)d_in[1];
  const float* Wq = (const float*)d_in[2];
  const float* bq = (const float*)d_in[3];
  const float* Wk = (const float*)d_in[4];
  const float* bk = (const float*)d_in[5];
  const float* Wv = (const float*)d_in[6];
  const float* bv = (const float*)d_in[7];
  const float* Wo = (const float*)d_in[8];
  const float* bo = (const float*)d_in[9];

  char* ws = (char*)d_ws;
  const size_t SLAB = (size_t)TOK * DD * 2;           // 32 MB bf16 slab
  unsigned short* qbuf  = (unsigned short*)(ws);            // Q; later aliased by attn out
  unsigned short* kbuf  = (unsigned short*)(ws + SLAB);     // K
  unsigned short* vtbuf = (unsigned short*)(ws + 2 * SLAB); // V transposed
  float*          part  = (float*)(ws + SLAB);              // aliases kbuf (dead by then)

  dim3 blk(256);
  qkv_gemm_kernel<<<dim3(DD / 128, TOK / 64, 3), blk, 0, stream>>>(
      x, z, Wq, bq, Wk, bk, Wv, bv, qbuf, kbuf, vtbuf);
  attn_kernel<<<dim3(4096), blk, 0, stream>>>(qbuf, kbuf, vtbuf, qbuf);
  out_proj_kernel<<<dim3(8, 8, 8), blk, 0, stream>>>(qbuf, Wo, part);
  reduce_kernel<<<dim3((512 * 512) / 256), blk, 0, stream>>>(part, bo, (float*)d_out);
}
